// GNNStructEncoder_206158430348
// MI455X (gfx1250) — matrix operand order repaired and checked by reading, not compiled, not run on hardware
//
#include <hip/hip_runtime.h>
#include <hip/hip_bf16.h>
#include <math.h>

typedef __attribute__((ext_vector_type(16))) _Float16 v16h;
typedef __attribute__((ext_vector_type(8)))  _Float16 v8h;
typedef __attribute__((ext_vector_type(8)))  float    v8f;

#define LAM1 0.01f
#define LAM2 0.001f
#define LAM3 0.0001f
#define S_NB 10
#define DD   64
#define IN0  128

#define WMMA_F16(a,b,c) __builtin_amdgcn_wmma_f32_16x16x32_f16(false,(a),false,(b),(short)0,(c),false,false)

// ---------- fragment loaders ----------

// A fragment (16x32 f16) from an LDS row-major f16 tile with leading dim `ld`.
// 16-bit A layout: lanes 0-15: row=lane, K = kt*32 + {0..7, 16..23}
//                  lanes 16-31: row=lane-16, K = kt*32 + {8..15, 24..31}
__device__ inline v16h ldsA(const _Float16* aT, int lane, int kt, int ld) {
  int row = lane & 15, half = lane >> 4;
  const _Float16* p = aT + row * ld + kt * 32 + half * 8;
  v8h lo = *(const v8h*)(p);
  v8h hi = *(const v8h*)(p + 16);
  v16h a;
#pragma unroll
  for (int i = 0; i < 8; ++i) { a[i] = lo[i]; a[i + 8] = hi[i]; }
  return a;
}

// B fragment (32x16 f16) from pre-packed weights in global memory.
// Packed layout: pw[((ct*KT+kt)*32 + lane)*16 + j] = W[kt*32 + (j>>3)*16 + (lane>>4)*8 + (j&7)][ct*16 + (lane&15)]
__device__ inline v16h ldB(const _Float16* __restrict__ pw, int lane, int ct, int kt, int KT) {
  const _Float16* p = pw + (((ct * KT + kt) << 9) + (lane << 4));
  v8h lo = *(const v8h*)(p);
  v8h hi = *(const v8h*)(p + 8);
  v16h b;
#pragma unroll
  for (int i = 0; i < 8; ++i) { b[i] = lo[i]; b[i + 8] = hi[i]; }
  return b;
}

// one 16-col output tile of a 64x64 layer: C = A(16x64) * W[:,ct*16:+16] + bias
__device__ inline v8f layer64_ct(v16h a0, v16h a1,
                                 const _Float16* __restrict__ pw,
                                 const float* __restrict__ bias,
                                 int lane, int ct) {
  float bv = bias[ct * 16 + (lane & 15)];
  v8f c;
#pragma unroll
  for (int r = 0; r < 8; ++r) c[r] = bv;
  c = WMMA_F16(a0, ldB(pw, lane, ct, 0, 2), c);
  c = WMMA_F16(a1, ldB(pw, lane, ct, 1, 2), c);
  return c;
}

// full 16x64 tile through a 64x64 layer, result (optionally relu'd) to LDS f16 tile
__device__ inline void layer64_lds(const _Float16* aT, _Float16* outT,
                                   const _Float16* __restrict__ pw,
                                   const float* __restrict__ bias,
                                   int lane, bool relu) {
  v16h a0 = ldsA(aT, lane, 0, 64);
  v16h a1 = ldsA(aT, lane, 1, 64);
  int col = lane & 15, rb = (lane >> 4) * 8;
#pragma unroll
  for (int ct = 0; ct < 4; ++ct) {
    v8f c = layer64_ct(a0, a1, pw, bias, lane, ct);
#pragma unroll
    for (int r = 0; r < 8; ++r) {
      float v = c[r];
      if (relu) v = fmaxf(v, 0.f);
      outT[(rb + r) * 64 + ct * 16 + col] = (_Float16)v;
    }
  }
  __builtin_amdgcn_wave_barrier();
}

// load a 16x64 f32 tile from global into an LDS f16 tile (one wave)
__device__ inline void load_tile16x64(const float* __restrict__ src, _Float16* aT,
                                      int lane, int n0) {
#pragma unroll
  for (int i = 0; i < 8; ++i) {
    int e = i * 32 + lane;          // 256 float4 chunks
    int row = e >> 4, c4 = (e & 15) * 4;
    float4 v = *(const float4*)(src + (size_t)(n0 + row) * 64 + c4);
    _Float16* q = aT + row * 64 + c4;
    q[0] = (_Float16)v.x; q[1] = (_Float16)v.y;
    q[2] = (_Float16)v.z; q[3] = (_Float16)v.w;
  }
  __builtin_amdgcn_wave_barrier();
}

// ---------- kernels ----------

// pack a (KT*32) x 64 f32 weight into B-fragment order (f16)
__global__ void k_pack(const float* __restrict__ W, _Float16* __restrict__ out, int KT) {
  int i = blockIdx.x * blockDim.x + threadIdx.x;
  int total = KT * 2048;            // 4 ct * KT * 32 lanes * 16
  if (i >= total) return;
  int j = i & 15, l = (i >> 4) & 31, g = i >> 9;
  int ct = g / KT, kt = g % KT;
  int k = kt * 32 + (j >> 3) * 16 + (l >> 4) * 8 + (j & 7);
  int n = ct * 16 + (l & 15);
  out[i] = (_Float16)W[k * 64 + n];
}

// h0 = x @ W0 + b0 ; hsum initialized to h0
__global__ void __launch_bounds__(256) k_h0(const float* __restrict__ x,
                                            const _Float16* __restrict__ pw,
                                            const float* __restrict__ b0,
                                            float* __restrict__ h0,
                                            float* __restrict__ hsum, int N) {
  __shared__ __align__(16) _Float16 sA[8][16 * IN0];
  int lane = threadIdx.x & 31, w = threadIdx.x >> 5;
  int tile = blockIdx.x * 8 + w;
  if (tile >= (N >> 4)) return;
  int n0 = tile * 16;
  _Float16* aT = sA[w];
#pragma unroll
  for (int i = 0; i < 16; ++i) {    // 512 float4 chunks of the 16x128 tile
    int e = i * 32 + lane;
    int row = e >> 5, c4 = (e & 31) * 4;
    float4 v = *(const float4*)(x + (size_t)(n0 + row) * IN0 + c4);
    _Float16* q = aT + row * IN0 + c4;
    q[0] = (_Float16)v.x; q[1] = (_Float16)v.y;
    q[2] = (_Float16)v.z; q[3] = (_Float16)v.w;
  }
  __builtin_amdgcn_wave_barrier();
  v16h a[4];
#pragma unroll
  for (int kt = 0; kt < 4; ++kt) a[kt] = ldsA(aT, lane, kt, IN0);
  int col = lane & 15, rb = (lane >> 4) * 8;
#pragma unroll
  for (int ct = 0; ct < 4; ++ct) {
    float bv = b0[ct * 16 + col];
    v8f c;
#pragma unroll
    for (int r = 0; r < 8; ++r) c[r] = bv;
#pragma unroll
    for (int kt = 0; kt < 4; ++kt) c = WMMA_F16(a[kt], ldB(pw, lane, ct, kt, 4), c);
#pragma unroll
    for (int r = 0; r < 8; ++r) {
      size_t off = (size_t)(n0 + rb + r) * 64 + ct * 16 + col;
      h0[off] = c[r];
      hsum[off] = c[r];
    }
  }
}

// hsum[dst] += h0[src] over all edges (segment_sum)
__global__ void k_agg(const int* __restrict__ ei, const float* __restrict__ h0,
                      float* __restrict__ hsum, int E) {
  int t = blockIdx.x * blockDim.x + threadIdx.x;
  if (t >= E * 16) return;
  int e = t >> 4, c = (t & 15) * 4;
  int s = ei[e], d = ei[E + e];
  float4 v = *(const float4*)(h0 + (size_t)s * 64 + c);
  float* q = hsum + (size_t)d * 64 + c;
  atomicAdd(q + 0, v.x); atomicAdd(q + 1, v.y);
  atomicAdd(q + 2, v.z); atomicAdd(q + 3, v.w);
}

// l1 = gin_mlp(hsum): relu(z@W1+b1)@W2+b2
__global__ void __launch_bounds__(256) k_gin(const float* __restrict__ hsum,
                                             const _Float16* __restrict__ p1, const float* __restrict__ b1,
                                             const _Float16* __restrict__ p2, const float* __restrict__ b2,
                                             float* __restrict__ l1, int N) {
  __shared__ __align__(16) _Float16 sA[8][16 * 64], sB[8][16 * 64];
  int lane = threadIdx.x & 31, w = threadIdx.x >> 5;
  int tile = blockIdx.x * 8 + w;
  if (tile >= (N >> 4)) return;
  int n0 = tile * 16;
  _Float16 *aT = sA[w], *bT = sB[w];
  load_tile16x64(hsum, aT, lane, n0);
  layer64_lds(aT, bT, p1, b1, lane, true);
  v16h a0 = ldsA(bT, lane, 0, 64), a1 = ldsA(bT, lane, 1, 64);
  int col = lane & 15, rb = (lane >> 4) * 8;
#pragma unroll
  for (int ct = 0; ct < 4; ++ct) {
    v8f c = layer64_ct(a0, a1, p2, b2, lane, ct);
#pragma unroll
    for (int r = 0; r < 8; ++r)
      l1[(size_t)(n0 + rb + r) * 64 + ct * 16 + col] = c[r];
  }
}

// degree branch: 4x relu(64x64) + 64->1 linear, relu, squared error vs gt
__global__ void __launch_bounds__(256) k_deg(const float* __restrict__ l1,
                                             const _Float16* p0, const float* c0,
                                             const _Float16* p1, const float* c1,
                                             const _Float16* p2, const float* c2,
                                             const _Float16* p3, const float* c3,
                                             const float* __restrict__ w5, const float* __restrict__ b5,
                                             const float* __restrict__ gt,
                                             float* __restrict__ out_dpn, int N) {
  __shared__ __align__(16) _Float16 sA[8][16 * 64], sB[8][16 * 64];
  int lane = threadIdx.x & 31, w = threadIdx.x >> 5;
  int tile = blockIdx.x * 8 + w;
  if (tile >= (N >> 4)) return;
  int n0 = tile * 16;
  _Float16 *aT = sA[w], *bT = sB[w];
  load_tile16x64(l1, aT, lane, n0);
  layer64_lds(aT, bT, p0, c0, lane, true);
  layer64_lds(bT, aT, p1, c1, lane, true);
  layer64_lds(aT, bT, p2, c2, lane, true);
  layer64_lds(bT, aT, p3, c3, lane, true);
  if (lane < 16) {
    float acc = b5[0];
#pragma unroll
    for (int d = 0; d < 64; ++d) acc += (float)aT[lane * 64 + d] * w5[d];
    float logit = fmaxf(acc, 0.f);
    float df = logit - gt[n0 + lane];
    out_dpn[n0 + lane] = df * df;
  }
}

// mean/sigma heads + feature branch (4-layer fnn) + per-node feature loss
__global__ void __launch_bounds__(256) k_featms(const float* __restrict__ l1,
                                                const float* __restrict__ h0,
                                                const _Float16* pm, const float* bm,
                                                const _Float16* ps, const float* bs,
                                                const _Float16* f0, const float* c0,
                                                const _Float16* f1, const float* c1,
                                                const _Float16* f2, const float* c2,
                                                const _Float16* f3, const float* c3,
                                                float* __restrict__ meanB,
                                                float* __restrict__ expsB,
                                                float* __restrict__ out_fpn, int N) {
  __shared__ __align__(16) _Float16 sA[8][16 * 64], sB[8][16 * 64];
  int lane = threadIdx.x & 31, w = threadIdx.x >> 5;
  int tile = blockIdx.x * 8 + w;
  if (tile >= (N >> 4)) return;
  int n0 = tile * 16;
  _Float16 *aT = sA[w], *bT = sB[w];
  load_tile16x64(l1, aT, lane, n0);
  int col = lane & 15, rb = (lane >> 4) * 8;
  v16h a0 = ldsA(aT, lane, 0, 64), a1 = ldsA(aT, lane, 1, 64);
#pragma unroll
  for (int ct = 0; ct < 4; ++ct) {          // mean head
    v8f c = layer64_ct(a0, a1, pm, bm, lane, ct);
#pragma unroll
    for (int r = 0; r < 8; ++r)
      meanB[(size_t)(n0 + rb + r) * 64 + ct * 16 + col] = c[r];
  }
#pragma unroll
  for (int ct = 0; ct < 4; ++ct) {          // sigma head, store exp(sigma)
    v8f c = layer64_ct(a0, a1, ps, bs, lane, ct);
#pragma unroll
    for (int r = 0; r < 8; ++r)
      expsB[(size_t)(n0 + rb + r) * 64 + ct * 16 + col] = expf(c[r]);
  }
  layer64_lds(aT, bT, f0, c0, lane, true);
  layer64_lds(bT, aT, f1, c1, lane, true);
  layer64_lds(aT, bT, f2, c2, lane, true);
  v16h g0 = ldsA(bT, lane, 0, 64), g1 = ldsA(bT, lane, 1, 64);
  float part[8];
#pragma unroll
  for (int r = 0; r < 8; ++r) part[r] = 0.f;
#pragma unroll
  for (int ct = 0; ct < 4; ++ct) {
    v8f c = layer64_ct(g0, g1, f3, c3, lane, ct);
#pragma unroll
    for (int r = 0; r < 8; ++r) {
      float h = h0[(size_t)(n0 + rb + r) * 64 + ct * 16 + col];
      float df = h - c[r];
      part[r] += df * df;
    }
  }
#pragma unroll
  for (int m = 8; m >= 1; m >>= 1)
#pragma unroll
    for (int r = 0; r < 8; ++r) part[r] += __shfl_xor(part[r], m, 32);
  if (col == 0) {
#pragma unroll
    for (int r = 0; r < 8; ++r)
      out_fpn[n0 + rb + r] = part[r] * (1.f / 64.f);
  }
}

// neighborhood statistics m2, v2 per (node, dim)
__global__ void k_stats(const float* __restrict__ h0, const int* __restrict__ nidx,
                        const float* __restrict__ nmask,
                        float* __restrict__ m2, float* __restrict__ v2, int N) {
  int t = blockIdx.x * blockDim.x + threadIdx.x;
  if (t >= N * DD) return;
  int n = t >> 6, d = t & 63;
  float vals[S_NB], msk[S_NB];
  float s = 0.f, lsum = 0.f;
#pragma unroll
  for (int k = 0; k < S_NB; ++k) {
    int id = nidx[n * S_NB + k];
    float m = nmask[n * S_NB + k];
    float v = m * h0[(size_t)id * 64 + d];
    vals[k] = v; msk[k] = m;
    s += v; lsum += m;
  }
  float L = fmaxf(lsum, 1.f);
  float mu = s / L;
  float q = 0.f;
#pragma unroll
  for (int k = 0; k < S_NB; ++k) {
    float dd = vals[k] - mu;
    q += msk[k] * dd * dd;
  }
  m2[t] = mu;
  v2[t] = q / L + 1.f;
}

// generator: per node, 3 samples of (N,S,64) through 4-layer fnn + KL vs (m2,v2)
__global__ void __launch_bounds__(256) k_gen(const float* __restrict__ meanB,
                                             const float* __restrict__ expsB,
                                             const float* __restrict__ stdz,
                                             const float* __restrict__ m2,
                                             const float* __restrict__ v2,
                                             const _Float16* g0w, const float* g0b,
                                             const _Float16* g1w, const float* g1b,
                                             const _Float16* g2w, const float* g2b,
                                             const _Float16* g3w, const float* g3b,
                                             float* __restrict__ out_hpn, int N) {
  __shared__ __align__(16) _Float16 sA[8][16 * 64], sB[8][16 * 64];
  int lane = threadIdx.x & 31, w = threadIdx.x >> 5;
  int n = blockIdx.x * 8 + w;
  if (n >= N) return;
  _Float16 *aT = sA[w], *bT = sB[w];
  int col = lane & 15;
  float mn0 = meanB[(size_t)n * 64 + lane];
  float mn1 = meanB[(size_t)n * 64 + lane + 32];
  float es0 = expsB[(size_t)n * 64 + lane];
  float es1 = expsB[(size_t)n * 64 + lane + 32];
  float m2v[4], v2v[4];
#pragma unroll
  for (int ct = 0; ct < 4; ++ct) {
    int d = ct * 16 + col;
    m2v[ct] = m2[(size_t)n * 64 + d];
    v2v[ct] = v2[(size_t)n * 64 + d];
  }
  float kl = 0.f;
  for (int t = 0; t < 3; ++t) {
    const float* z = stdz + (((size_t)t * N + n) * S_NB) * 64;
#pragma unroll
    for (int s = 0; s < S_NB; ++s) {
      aT[s * 64 + lane]      = (_Float16)(mn0 + es0 * z[s * 64 + lane]);
      aT[s * 64 + lane + 32] = (_Float16)(mn1 + es1 * z[s * 64 + lane + 32]);
    }
#pragma unroll
    for (int s = S_NB; s < 16; ++s) {
      aT[s * 64 + lane] = (_Float16)0.f;
      aT[s * 64 + lane + 32] = (_Float16)0.f;
    }
    __builtin_amdgcn_wave_barrier();
    layer64_lds(aT, bT, g0w, g0b, lane, true);
    layer64_lds(bT, aT, g1w, g1b, lane, true);
    layer64_lds(aT, bT, g2w, g2b, lane, true);
    v16h a0 = ldsA(bT, lane, 0, 64), a1 = ldsA(bT, lane, 1, 64);
#pragma unroll
    for (int ct = 0; ct < 4; ++ct) {
      v8f c = layer64_ct(a0, a1, g3w, g3b, lane, ct);
      // rows: lanes 0-15 hold rows 0-7, lanes 16-31 rows 8-15; only s<10 valid
      float sum = c[0] + c[1];
      float sq  = c[0] * c[0] + c[1] * c[1];
      if (lane < 16) {
#pragma unroll
        for (int r = 2; r < 8; ++r) { sum += c[r]; sq += c[r] * c[r]; }
      }
      sum += __shfl_xor(sum, 16, 32);
      sq  += __shfl_xor(sq, 16, 32);
      float m1 = sum * 0.1f;
      float v1 = sq * 0.1f - m1 * m1 + 1.f;
      if (lane < 16) {
        float md = m1 - m2v[ct];
        kl += logf(v2v[ct] / v1) + (v1 + md * md) / v2v[ct] - 1.f;
      }
    }
  }
#pragma unroll
  for (int m = 16; m >= 1; m >>= 1) kl += __shfl_xor(kl, m, 32);
  if (lane == 0) out_hpn[n] = kl * (0.5f / 3.f);
}

// combine per-node losses + global reduction (single block, deterministic)
__global__ void __launch_bounds__(256) k_final(float* __restrict__ out, int N) {
  __shared__ float sh[256], sd[256], sf[256];
  int tid = threadIdx.x;
  float ah = 0.f, ad = 0.f, af = 0.f;
  for (int n = tid; n < N; n += 256) {
    float h = out[1 + N + n];
    float d = out[1 + 2 * N + n];
    float f = out[1 + 3 * N + n];
    out[1 + n] = LAM1 * h + LAM3 * d + LAM2 * f;
    ah += h; ad += d; af += f;
  }
  sh[tid] = ah; sd[tid] = ad; sf[tid] = af;
  __syncthreads();
  for (int s = 128; s > 0; s >>= 1) {
    if (tid < s) { sh[tid] += sh[tid + s]; sd[tid] += sd[tid + s]; sf[tid] += sf[tid + s]; }
    __syncthreads();
  }
  if (tid == 0)
    out[0] = LAM1 * sh[0] + LAM3 * sd[0] / (float)N + LAM2 * sf[0] / (float)N;
}

// ---------- launch ----------

extern "C" void kernel_launch(void* const* d_in, const int* in_sizes, int n_in,
                              void* d_out, int out_size, void* d_ws, size_t ws_size,
                              hipStream_t stream) {
  const int N = in_sizes[2];          // gt_degree
  const int E = in_sizes[1] / 2;      // edge_index (2,E)
  const size_t NN = (size_t)N * 64;

  const float* x     = (const float*)d_in[0];
  const int*   ei    = (const int*)d_in[1];
  const float* gt    = (const float*)d_in[2];
  const int*   nidx  = (const int*)d_in[3];
  const float* nmask = (const float*)d_in[4];
  const float* stdz  = (const float*)d_in[5];

  float* h0    = (float*)d_ws;
  float* hsum  = h0 + NN;
  float* l1    = hsum + NN;
  float* meanB = l1 + NN;
  float* expsB = meanB + NN;
  float* m2    = expsB + NN;
  float* v2    = m2 + NN;
  _Float16* pk = (_Float16*)(v2 + NN);   // packed mlp0 (8192 halves)
  _Float16* pm = pk + 8192;              // 16 packed 64x64 mats, 4096 halves each

  // pack weights into B-fragment order
  k_pack<<<(8192 + 255) / 256, 256, 0, stream>>>((const float*)d_in[6], pk, 4);
  const int widx[16] = {8, 10, 12, 14, 16, 18, 20, 22, 24, 26, 28, 30, 34, 36, 38, 40};
  for (int i = 0; i < 16; ++i)
    k_pack<<<(4096 + 255) / 256, 256, 0, stream>>>((const float*)d_in[widx[i]],
                                                   pm + (size_t)i * 4096, 2);
  _Float16* gin0 = pm + 0 * 4096, * gin1 = pm + 1 * 4096;
  _Float16* wmn  = pm + 2 * 4096, * wsg  = pm + 3 * 4096;
  _Float16* gn0  = pm + 4 * 4096, * gn1  = pm + 5 * 4096;
  _Float16* gn2  = pm + 6 * 4096, * gn3  = pm + 7 * 4096;
  _Float16* dg0  = pm + 8 * 4096, * dg1  = pm + 9 * 4096;
  _Float16* dg2  = pm + 10 * 4096, * dg3 = pm + 11 * 4096;
  _Float16* ft0  = pm + 12 * 4096, * ft1 = pm + 13 * 4096;
  _Float16* ft2  = pm + 14 * 4096, * ft3 = pm + 15 * 4096;

  float* out   = (float*)d_out;
  float* o_hpn = out + 1 + (size_t)N;
  float* o_dpn = out + 1 + 2 * (size_t)N;
  float* o_fpn = out + 1 + 3 * (size_t)N;

  int tiles = N / 16;
  int gblocks = (tiles + 7) / 8;

  k_h0<<<gblocks, 256, 0, stream>>>(x, pk, (const float*)d_in[7], h0, hsum, N);
  k_agg<<<(E * 16 + 255) / 256, 256, 0, stream>>>(ei, h0, hsum, E);
  k_gin<<<gblocks, 256, 0, stream>>>(hsum, gin0, (const float*)d_in[9],
                                     gin1, (const float*)d_in[11], l1, N);
  k_deg<<<gblocks, 256, 0, stream>>>(l1,
      dg0, (const float*)d_in[25], dg1, (const float*)d_in[27],
      dg2, (const float*)d_in[29], dg3, (const float*)d_in[31],
      (const float*)d_in[32], (const float*)d_in[33], gt, o_dpn, N);
  k_featms<<<gblocks, 256, 0, stream>>>(l1, h0,
      wmn, (const float*)d_in[13], wsg, (const float*)d_in[15],
      ft0, (const float*)d_in[35], ft1, (const float*)d_in[37],
      ft2, (const float*)d_in[39], ft3, (const float*)d_in[41],
      meanB, expsB, o_fpn, N);
  k_stats<<<((size_t)N * 64 + 255) / 256, 256, 0, stream>>>(h0, nidx, nmask, m2, v2, N);
  k_gen<<<(N + 7) / 8, 256, 0, stream>>>(meanB, expsB, stdz, m2, v2,
      gn0, (const float*)d_in[17], gn1, (const float*)d_in[19],
      gn2, (const float*)d_in[21], gn3, (const float*)d_in[23],
      o_hpn, N);
  k_final<<<1, 256, 0, stream>>>(out, N);
}
